// DynamicBalancedMarginInnerProduct_61297773248760
// MI455X (gfx1250) — matrix operand order repaired and checked by reading, not compiled, not run on hardware
//
#include <hip/hip_runtime.h>
#include <hip/hip_bf16.h>

#define B_N 1024
#define D_K 512
#define C_N 100000
#define NTILES (C_N / 16)   // 6250
#define SCALE_F 30.0f

typedef __attribute__((ext_vector_type(16))) __bf16 v16bf;
typedef __attribute__((ext_vector_type(8)))  float  v8f;

union FragB16 {
    v16bf v;
    uint4 q[2];
    unsigned int u[8];
};

// ---------------------------------------------------------------------------
// Kernel 1: feature norms + normalized feat in split bf16 (hi/lo)
// grid: 1024 blocks x 256 threads
// ---------------------------------------------------------------------------
__global__ void knorm_feat(const float* __restrict__ feat,
                           unsigned short* __restrict__ nfh,
                           unsigned short* __restrict__ nfl,
                           float* __restrict__ normf) {
    __shared__ float red[256];
    __shared__ float s_inv;
    const int row = blockIdx.x;
    const int t = threadIdx.x;
    const float* fr = feat + (size_t)row * D_K;
    float x0 = fr[t];
    float x1 = fr[t + 256];
    red[t] = x0 * x0 + x1 * x1;
    __syncthreads();
    for (int s = 128; s > 0; s >>= 1) {
        if (t < s) red[t] += red[t + s];
        __syncthreads();
    }
    if (t == 0) {
        float nrm = sqrtf(red[0]);
        normf[row] = nrm;
        s_inv = 1.0f / nrm;
    }
    __syncthreads();
    const float inv = s_inv;
    const size_t base = (size_t)row * D_K;
    {
        float y = x0 * inv;
        unsigned int u = __float_as_uint(y);
        unsigned int h = u & 0xffff0000u;
        float r = y - __uint_as_float(h);
        nfh[base + t] = (unsigned short)(u >> 16);
        nfl[base + t] = (unsigned short)(__float_as_uint(r) >> 16);
    }
    {
        float y = x1 * inv;
        unsigned int u = __float_as_uint(y);
        unsigned int h = u & 0xffff0000u;
        float r = y - __uint_as_float(h);
        nfh[base + t + 256] = (unsigned short)(u >> 16);
        nfl[base + t + 256] = (unsigned short)(__float_as_uint(r) >> 16);
    }
}

// ---------------------------------------------------------------------------
// Kernel 2: class-weight norms + reciprocals
// grid: 100000 blocks x 128 threads (float4 per thread)
// ---------------------------------------------------------------------------
__global__ void knorm_w(const float* __restrict__ w,
                        float* __restrict__ normw,
                        float* __restrict__ invw) {
    __shared__ float red[128];
    const int row = blockIdx.x;
    const int t = threadIdx.x;
    const float4* wr = (const float4*)(w + (size_t)row * D_K);
    float4 v = wr[t];
    red[t] = v.x * v.x + v.y * v.y + v.z * v.z + v.w * v.w;
    __syncthreads();
    for (int s = 64; s > 0; s >>= 1) {
        if (t < s) red[t] += red[t + s];
        __syncthreads();
    }
    if (t == 0) {
        float n = sqrtf(red[0]);
        normw[row] = n;
        invw[row] = 1.0f / n;
    }
}

// ---------------------------------------------------------------------------
// Kernel 3: deterministic averages of the norms -> scalar outputs 4,5
// ---------------------------------------------------------------------------
__global__ void kavg(const float* __restrict__ normf,
                     const float* __restrict__ normw,
                     float* __restrict__ scal) {
    __shared__ double red[256];
    const int t = threadIdx.x;
    double s = 0.0;
    for (int i = t; i < B_N; i += 256) s += (double)normf[i];
    red[t] = s;
    __syncthreads();
    for (int k = 128; k > 0; k >>= 1) {
        if (t < k) red[t] += red[t + k];
        __syncthreads();
    }
    if (t == 0) scal[5] = (float)(red[0] / (double)B_N);  // avg_x_norm
    __syncthreads();
    s = 0.0;
    for (int i = t; i < C_N; i += 256) s += (double)normw[i];
    red[t] = s;
    __syncthreads();
    for (int k = 128; k > 0; k >>= 1) {
        if (t < k) red[t] += red[t + k];
        __syncthreads();
    }
    if (t == 0) scal[4] = (float)(red[0] / (double)C_N);  // avg_w_norm
}

// ---------------------------------------------------------------------------
// Kernel 4: cosine GEMM via split-bf16 WMMA.
// Block = 256 threads = 8 waves arranged 2(M) x 4(N); wave tile 64x32.
// Block tile 128(M) x 128(N).  grid = (1024/128, ceil(6250/8)) = (8, 782).
// cos = nf @ nw^T ; also writes SCALE*cos into marginal buffer.
// ---------------------------------------------------------------------------
__global__ __launch_bounds__(256) void kgemm(
    const unsigned short* __restrict__ nfh,
    const unsigned short* __restrict__ nfl,
    const float* __restrict__ w,
    const float* __restrict__ invw,
    float* __restrict__ cosOut,
    float* __restrict__ margOut) {
    const int tid  = threadIdx.x;
    const int lane = tid & 31;
    const int wv   = tid >> 5;     // 0..7
    const int wm   = wv >> 2;      // 0..1
    const int wn   = wv & 3;       // 0..3
    const int l15  = lane & 15;
    const int lh   = lane >> 4;    // 0 or 1

    const int mWave  = blockIdx.x * 128 + wm * 64;  // + mt*16
    const int nTile0 = blockIdx.y * 8 + wn * 2;     // 16-wide tile index

    // Hoisted per-N-tile info (each lane owns one weight row per tile).
    float  invn[2];
    size_t wbase[2];
    bool   valid[2];
#pragma unroll
    for (int nt = 0; nt < 2; ++nt) {
        int tN = nTile0 + nt;
        valid[nt] = (tN < NTILES);
        int r = tN * 16 + l15;
        if (r > C_N - 1) r = C_N - 1;  // clamp OOB loads
        wbase[nt] = (size_t)r * D_K;
        invn[nt] = invw[r];
    }

    size_t abase[4];
#pragma unroll
    for (int mt = 0; mt < 4; ++mt)
        abase[mt] = (size_t)(mWave + mt * 16 + l15) * D_K;

    v8f acc[4][2];
    const v8f vzero = {0.f, 0.f, 0.f, 0.f, 0.f, 0.f, 0.f, 0.f};
#pragma unroll
    for (int mt = 0; mt < 4; ++mt)
#pragma unroll
        for (int nt = 0; nt < 2; ++nt) acc[mt][nt] = vzero;

#pragma unroll 1
    for (int kk = 0; kk < D_K; kk += 32) {
        // --- A fragments (precomputed bf16 hi/lo, 16-bit A layout) ---
        // lane<16: K kk+0..7 then kk+16..23 ; lane>=16: K kk+8..15 then kk+24..31
        FragB16 ah[4], al[4];
        const int kA1 = kk + lh * 8;
#pragma unroll
        for (int mt = 0; mt < 4; ++mt) {
            ah[mt].q[0] = *(const uint4*)(nfh + abase[mt] + kA1);
            ah[mt].q[1] = *(const uint4*)(nfh + abase[mt] + kA1 + 16);
            al[mt].q[0] = *(const uint4*)(nfl + abase[mt] + kA1);
            al[mt].q[1] = *(const uint4*)(nfl + abase[mt] + kA1 + 16);
        }

        // --- B fragments on the fly: f32 row -> normalize -> split bf16 ---
        // 16-bit B layout: lane<16: K kk+0..7 then kk+16..23 ; lane>=16: K kk+8..15 then kk+24..31
        const int kB = kk + lh * 8;
#pragma unroll
        for (int nt = 0; nt < 2; ++nt) {
            const float4* wp = (const float4*)(w + wbase[nt] + kB);
            float4 w0 = wp[0], w1 = wp[1], w2 = wp[4], w3 = wp[5];
            float fv[16];
            fv[0] = w0.x; fv[1] = w0.y; fv[2]  = w0.z; fv[3]  = w0.w;
            fv[4] = w1.x; fv[5] = w1.y; fv[6]  = w1.z; fv[7]  = w1.w;
            fv[8] = w2.x; fv[9] = w2.y; fv[10] = w2.z; fv[11] = w2.w;
            fv[12] = w3.x; fv[13] = w3.y; fv[14] = w3.z; fv[15] = w3.w;
            FragB16 bh, bl;
            const float s = invn[nt];
#pragma unroll
            for (int j = 0; j < 8; ++j) {
                float y0 = fv[2 * j] * s;
                float y1 = fv[2 * j + 1] * s;
                unsigned int u0 = __float_as_uint(y0);
                unsigned int u1 = __float_as_uint(y1);
                unsigned int h0 = u0 & 0xffff0000u;
                unsigned int h1 = u1 & 0xffff0000u;
                float r0 = y0 - __uint_as_float(h0);
                float r1 = y1 - __uint_as_float(h1);
                bh.u[j] = (h0 >> 16) | h1;
                bl.u[j] = (__float_as_uint(r0) >> 16) |
                          (__float_as_uint(r1) & 0xffff0000u);
            }
#pragma unroll
            for (int mt = 0; mt < 4; ++mt) {
                acc[mt][nt] = __builtin_amdgcn_wmma_f32_16x16x32_bf16(
                    false, ah[mt].v, false, bh.v, (short)0, acc[mt][nt], false, false);
                acc[mt][nt] = __builtin_amdgcn_wmma_f32_16x16x32_bf16(
                    false, ah[mt].v, false, bl.v, (short)0, acc[mt][nt], false, false);
                acc[mt][nt] = __builtin_amdgcn_wmma_f32_16x16x32_bf16(
                    false, al[mt].v, false, bh.v, (short)0, acc[mt][nt], false, false);
            }
        }
    }

    // --- store C/D: VGPR r -> M = lh*8 + r, N = l15 within tile ---
#pragma unroll
    for (int nt = 0; nt < 2; ++nt) {
        if (!valid[nt]) continue;
        const int gn = (nTile0 + nt) * 16 + l15;
#pragma unroll
        for (int mt = 0; mt < 4; ++mt) {
            const int mBase = mWave + mt * 16 + lh * 8;
#pragma unroll
            for (int r = 0; r < 8; ++r) {
                size_t idx = (size_t)(mBase + r) * C_N + gn;
                float c = acc[mt][nt][r];
                cosOut[idx] = c;
                margOut[idx] = SCALE_F * c;
            }
        }
    }
}

// ---------------------------------------------------------------------------
// Kernel 5: theta statistics + dynamic-margin fixup of target logits.
// One block, 1024 threads (one per sample).
// ---------------------------------------------------------------------------
__global__ __launch_bounds__(1024) void kstats(
    const int* __restrict__ label,
    const float* __restrict__ cosOut,
    float* __restrict__ margOut,
    float* __restrict__ scal) {
    __shared__ float sMax[1024];
    __shared__ float sMin[1024];
    __shared__ double sSum[1024];
    __shared__ double sSq[1024];
    __shared__ float bAvg, bMin, bMax;

    const int i = threadIdx.x;
    const int li = label[i];
    const size_t idx = (size_t)i * C_N + (size_t)li;
    const float c = cosOut[idx];
    const float ct = fminf(fmaxf(c, -1.0f), 1.0f);
    const float theta = acosf(ct) * 57.29577951308232f;  // degrees

    sMax[i] = theta;
    sMin[i] = theta;
    sSum[i] = (double)theta;
    sSq[i] = (double)theta * (double)theta;
    __syncthreads();
    for (int s = 512; s > 0; s >>= 1) {
        if (i < s) {
            sMax[i] = fmaxf(sMax[i], sMax[i + s]);
            sMin[i] = fminf(sMin[i], sMin[i + s]);
            sSum[i] += sSum[i + s];
            sSq[i] += sSq[i + s];
        }
        __syncthreads();
    }
    if (i == 0) {
        const double n = (double)B_N;
        double sum = sSum[0], sq = sSq[0];
        float avg = (float)(sum / n);
        double var = (sq - sum * sum / n) / (n - 1.0);
        if (var < 0.0) var = 0.0;
        bAvg = avg;
        bMin = sMin[0];
        bMax = sMax[0];
        scal[0] = avg;               // avg_theta
        scal[1] = sMin[0];           // min_theta
        scal[2] = sMax[0];           // max_theta
        scal[3] = (float)sqrt(var);  // stdv_theta (ddof=1)
    }
    __syncthreads();

    const float avg = bAvg, mn = bMin, mx = bMax;
    const float D2R = 0.017453292519943295f;
    const float mAbove = (mx < 90.0f) ? (90.0f - avg) * D2R : 0.0f;
    const float mBelow = mn * D2R;
    const float mrg = (theta > avg) ? mAbove : mBelow;
    const bool inr = ct > -cosf(mrg);
    const float madd = inr ? mrg : 0.0f;
    const float mext = inr ? 0.0f : (-mrg * sinf(mrg));
    const float tm = cosf(acosf(ct) + madd) + mext;
    margOut[idx] = SCALE_F * tm;
}

// ---------------------------------------------------------------------------
extern "C" void kernel_launch(void* const* d_in, const int* in_sizes, int n_in,
                              void* d_out, int out_size, void* d_ws, size_t ws_size,
                              hipStream_t stream) {
    const float* feat = (const float*)d_in[0];
    const int* label = (const int*)d_in[1];
    const float* w = (const float*)d_in[2];

    float* out = (float*)d_out;
    float* cosOut = out;
    float* margOut = out + (size_t)B_N * C_N;
    float* scal = out + 2 * (size_t)B_N * C_N;

    char* ws = (char*)d_ws;
    float* normf = (float*)ws;                                     // 1024 f32
    float* normw = (float*)(ws + 4096);                            // 100000 f32
    float* invw = (float*)(ws + 4096 + 400000);                    // 100000 f32
    unsigned short* nfh = (unsigned short*)(ws + 4096 + 800000);   // 1024*512 bf16
    unsigned short* nfl = nfh + (size_t)B_N * D_K;                 // 1024*512 bf16

    knorm_feat<<<B_N, 256, 0, stream>>>(feat, nfh, nfl, normf);
    knorm_w<<<C_N, 128, 0, stream>>>(w, normw, invw);
    kavg<<<1, 256, 0, stream>>>(normf, normw, scal);
    kgemm<<<dim3(8, (NTILES + 7) / 8), 256, 0, stream>>>(nfh, nfl, w, invw,
                                                         cosOut, margOut);
    kstats<<<1, 1024, 0, stream>>>(label, cosOut, margOut, scal);
}
